// SwinTransformerBlock_76467597738326
// MI455X (gfx1250) — matrix operand order repaired and checked by reading, not compiled, not run on hardware
//
#include <hip/hip_runtime.h>
#include <hip/hip_bf16.h>

// ---------------------------------------------------------------------------
// Swin block on gfx1250: bf16 WMMA (v_wmma_f32_16x16x32_bf16) for every GEMM.
// Round 3: double-buffered A *and* B fragments in the streaming GEMMs so each
// WMMA consumes operands loaded one K-step earlier (no full loadcnt drains).
// ---------------------------------------------------------------------------

#define HH 56
#define WW_ 56
#define WS 7
#define SHIFT 3
#define HEADS 12
#define CC 384
#define HD 32
#define LW 49
#define NTOK (2048 * 49)            // real tokens (B*H*W)
#define NTOKP (2048 * 64)           // padded window rows
#define SCALE_F 0.17677669529663689f

typedef __attribute__((ext_vector_type(16))) __bf16          v16bf;
typedef __attribute__((ext_vector_type(8)))  float           v8f;
typedef __attribute__((ext_vector_type(16))) unsigned short  u16x16;
typedef __attribute__((ext_vector_type(8)))  unsigned short  u16x8;

__device__ __forceinline__ unsigned short f2bf(float f) {
  unsigned int u = __builtin_bit_cast(unsigned int, f);
  u += 0x7FFFu + ((u >> 16) & 1u);          // round-to-nearest-even
  return (unsigned short)(u >> 16);
}

__device__ __forceinline__ v8f wmma_bf16(v16bf a, v16bf b, v8f c) {
  return __builtin_amdgcn_wmma_f32_16x16x32_bf16(false, a, false, b,
                                                 (short)0, c, false, false);
}

// A fragment: 16x32 bf16 tile, row-major source (row stride in elements).
// Branchless: caller guarantees rows m0..m0+15 readable (padded buffers).
__device__ __forceinline__ v16bf load_A_frag(const unsigned short* base,
                                             int row_stride, int m0, int k0) {
  int lane = threadIdx.x & 31;
  int half = lane >> 4;
  int m = m0 + (lane & 15);
  const unsigned short* p = base + (size_t)m * row_stride + k0 + half * 16;
  u16x8 lo = *(const u16x8*)p;
  u16x8 hi = *(const u16x8*)(p + 8);
  u16x16 r = __builtin_shufflevector(lo, hi, 0, 1, 2, 3, 4, 5, 6, 7,
                                     8, 9, 10, 11, 12, 13, 14, 15);
  return __builtin_bit_cast(v16bf, r);
}

// B fragment: p points at 16 contiguous K values for this lane (32B aligned).
__device__ __forceinline__ v16bf load_B_frag(const unsigned short* p) {
  return __builtin_bit_cast(v16bf, *(const u16x16*)p);
}

// --------------------------- weight repack ---------------------------------
// fp32 W[K][N] -> bf16 Wt[K/32][N][32]
__global__ void k_wconv(const float* __restrict__ w,
                        unsigned short* __restrict__ wt, int K, int N) {
  int idx = blockIdx.x * 256 + threadIdx.x;
  if (idx >= K * N) return;
  int k = idx / N, n = idx - k * N;
  wt[(((size_t)(k >> 5)) * N + n) * 32 + (k & 31)] = f2bf(w[idx]);
}

// --------- LN0 + cyclic shift + window partition (64-row padded) -----------
__global__ __launch_bounds__(256) void k_ln0(const float* __restrict__ x,
                                             const float* __restrict__ g,
                                             const float* __restrict__ bt,
                                             unsigned short* __restrict__ hwin) {
  int tokId = blockIdx.x * 8 + (threadIdx.x >> 5);   // 8 rows / block
  int lane = threadIdx.x & 31;
  int win = tokId >> 6, tok = tokId & 63;
  unsigned short* op = hwin + (size_t)tokId * CC;
  if (tok >= LW) {                                    // zero pad rows
#pragma unroll
    for (int u = 0; u < 12; ++u) op[lane + u * 32] = 0;
    return;
  }
  int b = win >> 6, wq = win & 63, wh = wq >> 3, ww = wq & 7;
  int i = tok / WS, j = tok - i * WS;
  int row = (wh * WS + i + SHIFT) % HH;
  int col = (ww * WS + j + SHIFT) % WW_;
  const float* xp = x + ((size_t)b * (HH * WW_) + row * WW_ + col) * CC;

  float v[12];
  float s = 0.f;
#pragma unroll
  for (int u = 0; u < 12; ++u) { v[u] = xp[lane + u * 32]; s += v[u]; }
#pragma unroll
  for (int o = 16; o; o >>= 1) s += __shfl_xor(s, o, 32);
  float mean = s * (1.f / CC);
  float q = 0.f;
#pragma unroll
  for (int u = 0; u < 12; ++u) { float d = v[u] - mean; q += d * d; }
#pragma unroll
  for (int o = 16; o; o >>= 1) q += __shfl_xor(q, o, 32);
  float inv = rsqrtf(q * (1.f / CC) + 1e-5f);
#pragma unroll
  for (int u = 0; u < 12; ++u) {
    int c = lane + u * 32;
    op[c] = f2bf((v[u] - mean) * inv * g[c] + bt[c]);
  }
}

// --------------------------- LN1 (plain rows) ------------------------------
__global__ __launch_bounds__(256) void k_ln1(const float* __restrict__ x1,
                                             const float* __restrict__ g,
                                             const float* __restrict__ bt,
                                             unsigned short* __restrict__ hln) {
  int m = blockIdx.x * 8 + (threadIdx.x >> 5);
  int lane = threadIdx.x & 31;
  const float* xp = x1 + (size_t)m * CC;
  float v[12];
  float s = 0.f;
#pragma unroll
  for (int u = 0; u < 12; ++u) { v[u] = xp[lane + u * 32]; s += v[u]; }
#pragma unroll
  for (int o = 16; o; o >>= 1) s += __shfl_xor(s, o, 32);
  float mean = s * (1.f / CC);
  float q = 0.f;
#pragma unroll
  for (int u = 0; u < 12; ++u) { float d = v[u] - mean; q += d * d; }
#pragma unroll
  for (int o = 16; o; o >>= 1) q += __shfl_xor(q, o, 32);
  float inv = rsqrtf(q * (1.f / CC) + 1e-5f);
  unsigned short* op = hln + (size_t)m * CC;
#pragma unroll
  for (int u = 0; u < 12; ++u) {
    int c = lane + u * 32;
    op[c] = f2bf((v[u] - mean) * inv * g[c] + bt[c]);
  }
}

// ------------- fused QKV + attention per (window, head) --------------------
__global__ __launch_bounds__(128) void k_attn(
    const unsigned short* __restrict__ hwin,    // [2048][64][384] padded
    const unsigned short* __restrict__ wqkvt,   // [12][1152][32]
    const float* __restrict__ rpb,              // [169][12]
    const float* __restrict__ amask,            // [64][49][49]
    unsigned short* __restrict__ hout) {        // [2048][64][384] padded
  __shared__ __align__(16) unsigned short qs[64 * 32];
  __shared__ __align__(16) unsigned short ks[64 * 32];
  __shared__ __align__(16) unsigned short vs[32 * 64];
  __shared__ __align__(16) unsigned short ps[64 * 64];
  __shared__ __align__(16) float ss[64 * 64];

  int bid = blockIdx.x;
  int win = bid / HEADS, h = bid - win * HEADS;
  int wave = threadIdx.x >> 5;
  int lane = threadIdx.x & 31;
  int half = lane >> 4, ln = lane & 15;
  const unsigned short* aBase = hwin + (size_t)win * 64 * CC;

  // ---- stage 1: QKV_h = Xwin(64x384) @ Wqkv[:, head cols] ----
  // wave = M-tile; 6 N-accumulators (Q0 Q1 K0 K1 V0 V1); A loaded once per kt.
  {
    int mt = wave;
    v8f acc[6];
#pragma unroll
    for (int nt = 0; nt < 6; ++nt) acc[nt] = (v8f){};
    v16bf a = load_A_frag(aBase, CC, mt * 16, 0);
#pragma unroll
    for (int kt = 0; kt < 12; ++kt) {
      v16bf an = a;
      if (kt < 11) an = load_A_frag(aBase, CC, mt * 16, (kt + 1) * 32);
#pragma unroll
      for (int nt = 0; nt < 6; ++nt) {
        int n = (nt >> 1) * CC + h * HD + (nt & 1) * 16 + ln;
        v16bf b = load_B_frag(wqkvt + (((size_t)kt * 1152 + n) * 32 + half * 16));
        acc[nt] = wmma_bf16(a, b, acc[nt]);
      }
      a = an;
    }
#pragma unroll
    for (int nt = 0; nt < 6; ++nt) {
      int mat = nt >> 1;
      int c0 = (nt & 1) * 16 + ln;
#pragma unroll
      for (int r = 0; r < 8; ++r) {
        int tokr = mt * 16 + r + 8 * half;
        unsigned short bv = f2bf(acc[nt][r]);
        if (mat == 0) qs[tokr * 32 + c0] = bv;
        else if (mat == 1) ks[tokr * 32 + c0] = bv;
        else vs[c0 * 64 + tokr] = bv;              // V stored transposed
      }
    }
  }
  __syncthreads();

  // ---- stage 2: S = Q @ K^T  (wave = M-tile, 4 N-tiles, K=32) ----
  {
    int mt = wave;
    v16bf a = load_A_frag(qs, 32, mt * 16, 0);
#pragma unroll
    for (int nt = 0; nt < 4; ++nt) {
      int ntok = nt * 16 + ln;
      v16bf b = load_B_frag(ks + (ntok * 32 + half * 16));
      v8f acc = {};
      acc = wmma_bf16(a, b, acc);
#pragma unroll
      for (int r = 0; r < 8; ++r)
        ss[(mt * 16 + r + 8 * half) * 64 + ntok] = acc[r];
    }
  }
  __syncthreads();

  // ---- stage 3: softmax rows (scale + rel-pos bias + shift mask) ----
  if (threadIdx.x < 64) {
    int i = threadIdx.x;
    if (i < LW) {
      int nw = win & 63;
      int qi = i / WS, qj = i - qi * WS;
      float mx = -1e30f;
      for (int j = 0; j < LW; ++j) {
        int ki = j / WS, kj = j - ki * WS;
        int ridx = (qi - ki + WS - 1) * (2 * WS - 1) + (qj - kj + WS - 1);
        float sv = ss[i * 64 + j] * SCALE_F + rpb[ridx * HEADS + h] +
                   amask[((size_t)nw * LW + i) * LW + j];
        ss[i * 64 + j] = sv;
        mx = fmaxf(mx, sv);
      }
      float sum = 0.f;
      for (int j = 0; j < LW; ++j) {
        float e = __expf(ss[i * 64 + j] - mx);
        ss[i * 64 + j] = e;
        sum += e;
      }
      float inv = 1.f / sum;
      for (int j = 0; j < LW; ++j) ps[i * 64 + j] = f2bf(ss[i * 64 + j] * inv);
      for (int j = LW; j < 64; ++j) ps[i * 64 + j] = 0;
    } else {
      for (int j = 0; j < 64; ++j) ps[i * 64 + j] = 0;
    }
  }
  __syncthreads();

  // ---- stage 4: O = P @ V  (wave = M-tile, 2 N-accs, K=64) ----
  {
    int mt = wave;
    v8f acc[2] = {{}, {}};
#pragma unroll
    for (int kt = 0; kt < 2; ++kt) {
      v16bf a = load_A_frag(ps, 64, mt * 16, kt * 32);
#pragma unroll
      for (int nt = 0; nt < 2; ++nt) {
        v16bf b = load_B_frag(vs + ((nt * 16 + ln) * 64 + kt * 32 + half * 16));
        acc[nt] = wmma_bf16(a, b, acc[nt]);
      }
    }
#pragma unroll
    for (int nt = 0; nt < 2; ++nt)
#pragma unroll
      for (int r = 0; r < 8; ++r) {
        int tok = mt * 16 + r + 8 * half;
        hout[((size_t)win * 64 + tok) * CC + h * HD + nt * 16 + ln] =
            f2bf(acc[nt][r]);                      // pad rows store zeros
      }
  }
}

// ---------------------------------------------------------------------------
// Generic double-buffered 64xK @ Kx64 wave body: wave = M-tile, 4 N-accs.
// aBase: padded row-major bf16 A.  wt: [K/32][N][32] bf16 B panels.
// KT = K/32 steps.  Fully unrolled; both A and B are one K-step ahead.
// ---------------------------------------------------------------------------
template <int KT, int NDIM>
__device__ __forceinline__ void gemm_wave_64x64(
    const unsigned short* aBase, int aStride, const unsigned short* wt,
    int nBase, int mt, int half, int ln, v8f acc[4]) {
  v16bf a0 = load_A_frag(aBase, aStride, mt * 16, 0);
  v16bf b0[4];
#pragma unroll
  for (int s = 0; s < 4; ++s)
    b0[s] = load_B_frag(wt + (((size_t)0 * NDIM + nBase + s * 16 + ln) * 32 +
                              half * 16));
#pragma unroll
  for (int kt = 0; kt < KT - 1; ++kt) {
    v16bf a1 = load_A_frag(aBase, aStride, mt * 16, (kt + 1) * 32);
    v16bf b1[4];
#pragma unroll
    for (int s = 0; s < 4; ++s)
      b1[s] = load_B_frag(wt + (((size_t)(kt + 1) * NDIM + nBase + s * 16 + ln) *
                                    32 + half * 16));
#pragma unroll
    for (int s = 0; s < 4; ++s) acc[s] = wmma_bf16(a0, b0[s], acc[s]);
    a0 = a1;
#pragma unroll
    for (int s = 0; s < 4; ++s) b0[s] = b1[s];
  }
#pragma unroll
  for (int s = 0; s < 4; ++s) acc[s] = wmma_bf16(a0, b0[s], acc[s]);
}

// -------- proj GEMM + residual + window reverse + un-shift -> x1 -----------
__global__ __launch_bounds__(128) void k_proj(
    const unsigned short* __restrict__ hout,    // [2048][64][384] padded
    const unsigned short* __restrict__ wprojt,  // [12][384][32]
    const float* __restrict__ x, float* __restrict__ xout) {
  int win = blockIdx.y, nb = blockIdx.x;        // nb < 6
  int wave = threadIdx.x >> 5;
  int lane = threadIdx.x & 31;
  int half = lane >> 4, ln = lane & 15;
  const unsigned short* aBase = hout + (size_t)win * 64 * CC;
  int b = win >> 6, wq = win & 63, wh = wq >> 3, ww = wq & 7;
  int mt = wave;

  v8f acc[4];
#pragma unroll
  for (int s = 0; s < 4; ++s) acc[s] = (v8f){};
  gemm_wave_64x64<12, CC>(aBase, CC, wprojt, nb * 64, mt, half, ln, acc);

#pragma unroll
  for (int s = 0; s < 4; ++s) {
    int n = nb * 64 + s * 16 + ln;
#pragma unroll
    for (int r = 0; r < 8; ++r) {
      int tok = mt * 16 + r + 8 * half;
      if (tok < LW) {
        int i = tok / WS, j = tok - i * WS;
        int row = (wh * WS + i + SHIFT) % HH;
        int col = (ww * WS + j + SHIFT) % WW_;
        size_t off = ((size_t)b * (HH * WW_) + row * WW_ + col) * CC + n;
        xout[off] = x[off] + acc[s][r];
      }
    }
  }
}

// ----------------------- FFN layer 0: GEMM + GELU --------------------------
__global__ __launch_bounds__(128) void k_ffn0(
    const unsigned short* __restrict__ hln,
    const unsigned short* __restrict__ w0t,     // [12][1536][32]
    const float* __restrict__ bias0,
    unsigned short* __restrict__ g, int mOff) {
  int mb = blockIdx.y, nb = blockIdx.x;         // nb < 24
  int wave = threadIdx.x >> 5;
  int lane = threadIdx.x & 31;
  int half = lane >> 4, ln = lane & 15;
  const unsigned short* aBase = hln + (size_t)(mOff + mb * 64) * CC;
  int mt = wave;

  v8f acc[4];
#pragma unroll
  for (int s = 0; s < 4; ++s) acc[s] = (v8f){};
  gemm_wave_64x64<12, 1536>(aBase, CC, w0t, nb * 64, mt, half, ln, acc);

#pragma unroll
  for (int s = 0; s < 4; ++s) {
    int n = nb * 64 + s * 16 + ln;
    float bv = bias0[n];
#pragma unroll
    for (int r = 0; r < 8; ++r) {
      int m = mb * 64 + mt * 16 + r + 8 * half;     // slab-local row
      float v = acc[s][r] + bv;
      float ge = 0.5f * v * (1.f + erff(v * 0.70710678118654752f));
      g[(size_t)m * 1536 + n] = f2bf(ge);
    }
  }
}

// ------------------ FFN layer 1: GEMM + bias + residual --------------------
__global__ __launch_bounds__(128) void k_ffn1(
    const unsigned short* __restrict__ g,
    const unsigned short* __restrict__ w1t,     // [48][384][32]
    const float* __restrict__ bias1,
    float* __restrict__ out, int mOff) {
  int mb = blockIdx.y, nb = blockIdx.x;         // nb < 6
  int wave = threadIdx.x >> 5;
  int lane = threadIdx.x & 31;
  int half = lane >> 4, ln = lane & 15;
  const unsigned short* aBase = g + (size_t)mb * 64 * 1536;
  int mt = wave;

  v8f acc[4];
#pragma unroll
  for (int s = 0; s < 4; ++s) acc[s] = (v8f){};
  gemm_wave_64x64<48, CC>(aBase, 1536, w1t, nb * 64, mt, half, ln, acc);

#pragma unroll
  for (int s = 0; s < 4; ++s) {
    int n = nb * 64 + s * 16 + ln;
    float bv = bias1[n];
#pragma unroll
    for (int r = 0; r < 8; ++r) {
      int m = mOff + mb * 64 + mt * 16 + r + 8 * half;
      size_t off = (size_t)m * CC + n;
      out[off] = out[off] + acc[s][r] + bv;         // x1 + ffn
    }
  }
}

// ---------------------------------------------------------------------------
extern "C" void kernel_launch(void* const* d_in, const int* in_sizes, int n_in,
                              void* d_out, int out_size, void* d_ws,
                              size_t ws_size, hipStream_t stream) {
  (void)in_sizes; (void)n_in; (void)out_size; (void)ws_size;
  const float* x       = (const float*)d_in[0];
  const float* amask   = (const float*)d_in[1];
  const float* gamma0  = (const float*)d_in[2];
  const float* beta0   = (const float*)d_in[3];
  const float* w_qkv   = (const float*)d_in[4];
  const float* rpb     = (const float*)d_in[5];
  const float* w_proj  = (const float*)d_in[6];
  const float* gamma1  = (const float*)d_in[7];
  const float* beta1   = (const float*)d_in[8];
  const float* w_ffn0  = (const float*)d_in[9];
  const float* b_ffn0  = (const float*)d_in[10];
  const float* w_ffn1  = (const float*)d_in[11];
  const float* b_ffn1  = (const float*)d_in[12];
  float* out = (float*)d_out;

  char* base = (char*)d_ws;
  size_t off = 0;
  auto take = [&](size_t bytes) {
    char* p = base + off;
    off = (off + bytes + 255) & ~(size_t)255;
    return p;
  };
  unsigned short* wqkvt  = (unsigned short*)take((size_t)CC * 1152 * 2);
  unsigned short* wprojt = (unsigned short*)take((size_t)CC * CC * 2);
  unsigned short* wffn0t = (unsigned short*)take((size_t)CC * 1536 * 2);
  unsigned short* wffn1t = (unsigned short*)take((size_t)1536 * CC * 2);
  unsigned short* hwin   = (unsigned short*)take((size_t)NTOKP * CC * 2);
  unsigned short* hout   = (unsigned short*)take((size_t)NTOKP * CC * 2);
  unsigned short* hln    = (unsigned short*)take((size_t)NTOK * CC * 2);
  const int MSLAB = NTOK / 8;                        // 12544 rows per slab
  unsigned short* gbuf   = (unsigned short*)take((size_t)MSLAB * 1536 * 2);

  // weight repack (fp32 -> bf16 K-panel layout)
  k_wconv<<<(CC * 1152 + 255) / 256, 256, 0, stream>>>(w_qkv, wqkvt, CC, 1152);
  k_wconv<<<(CC * CC + 255) / 256, 256, 0, stream>>>(w_proj, wprojt, CC, CC);
  k_wconv<<<(CC * 1536 + 255) / 256, 256, 0, stream>>>(w_ffn0, wffn0t, CC, 1536);
  k_wconv<<<(1536 * CC + 255) / 256, 256, 0, stream>>>(w_ffn1, wffn1t, 1536, CC);

  // LN0 + shift + window partition into padded 64-row windows
  k_ln0<<<NTOKP / 8, 256, 0, stream>>>(x, gamma0, beta0, hwin);

  // fused QKV + attention: one block per (window, head)
  k_attn<<<2048 * HEADS, 128, 0, stream>>>(hwin, wqkvt, rpb, amask, hout);

  // proj + residual + window reverse -> x1 in d_out
  k_proj<<<dim3(6, 2048), 128, 0, stream>>>(hout, wprojt, x, out);

  // LN1
  k_ln1<<<NTOK / 8, 256, 0, stream>>>(out, gamma1, beta1, hln);

  // FFN in 8 slabs (reuse gbuf)
  for (int sIdx = 0; sIdx < 8; ++sIdx) {
    int mOff = sIdx * MSLAB;
    k_ffn0<<<dim3(24, MSLAB / 64), 128, 0, stream>>>(hln, wffn0t, b_ffn0, gbuf, mOff);
    k_ffn1<<<dim3(6, MSLAB / 64), 128, 0, stream>>>(gbuf, wffn1t, b_ffn1, out, mOff);
  }
}
